// VCReg_45964740002261
// MI455X (gfx1250) — matrix operand order repaired and checked by reading, not compiled, not run on hardware
//
#include <hip/hip_runtime.h>

// VCReg loss: B=C=64, HW=50176 (derived from in_sizes), f32 input, scalar f32 out.
//   std_loss: per-(b,h) unbiased std over 64 channels, mean(relu(1-std))
//   cov_loss: 64x64 covariance over (b,h);  cov*HW*(B-1) = sum yy^T - (1/B) sum_h s s^T
// Strategy: one streaming pass; Gram via bf16 WMMA (f32 accumulate), std in f32 VALU.

#define HC       128          // pixels per block chunk
#define LDSPITCH 136          // 128 + 8 pad (keeps 16B row alignment, shifts banks)
#define NB       64
#define NC       64

typedef __attribute__((ext_vector_type(16))) __bf16         v16bf;
typedef __attribute__((ext_vector_type(8)))  float          v8f;
typedef __attribute__((ext_vector_type(8)))  unsigned short u16x8;

union FragU { v16bf v; u16x8 h[2]; };

__device__ __forceinline__ unsigned short f2bf_rne(float x) {
    unsigned u = __float_as_uint(x);
    u += 0x7fffu + ((u >> 16) & 1u);      // round-to-nearest-even
    return (unsigned short)(u >> 16);
}

// A fragment (16x32 bf16): lane<16 -> M=lane, K in {0..7}U{16..23}; lane>=16 -> M=lane-16, K+8.
// Row-major [channel][k] LDS tile means both runs are contiguous: two b128 loads.
__device__ __forceinline__ v16bf load_fragA(const unsigned short* tile, int ci, int ks, int lane) {
    int m  = lane & 15;
    int k0 = ks * 32 + ((lane & 16) ? 8 : 0);
    const unsigned short* row = tile + (ci * 16 + m) * LDSPITCH;
    FragU f;
    f.h[0] = *(const u16x8*)(row + k0);
    f.h[1] = *(const u16x8*)(row + k0 + 16);
    return f.v;
}

// B fragment (32x16 bf16): lane<16 -> N=lane, K in {0..7}U{16..23}; lane>=16 -> N=lane-16, K+8.
__device__ __forceinline__ v16bf load_fragB(const unsigned short* tile, int di, int ks, int lane) {
    int n  = lane & 15;
    int k0 = ks * 32 + ((lane & 16) ? 8 : 0);
    const unsigned short* row = tile + (di * 16 + n) * LDSPITCH;
    FragU f;
    f.h[0] = *(const u16x8*)(row + k0);
    f.h[1] = *(const u16x8*)(row + k0 + 16);
    return f.v;
}

__global__ void __launch_bounds__(128)
vcreg_main(const float* __restrict__ Y, unsigned long long HW,
           float* __restrict__ gG, float* __restrict__ gStd) {
    __shared__ alignas(16) unsigned short ybf[NC * LDSPITCH];  // bf16 tile [64ch][128k]
    __shared__ float s_lds[NC * HC];                           // batch-sum accumulator
    __shared__ float red[128];

    const int t    = threadIdx.x;
    const int lane = t & 31;
    const int w    = t >> 5;                       // wave id == ci (owns tile row)
    const unsigned long long h0 = (unsigned long long)blockIdx.x * HC;

    for (int f = t; f < NC * HC; f += 128) s_lds[f] = 0.f;

    v8f accY[4], accS[4];
    #pragma unroll
    for (int di = 0; di < 4; ++di) { accY[di] = 0.f; accS[di] = 0.f; }
    float stdacc = 0.f;

    __syncthreads();

    for (int b = 0; b < NB; ++b) {
        // ---- phase 1: coalesced load (thread t owns pixel h0+t), stats + bf16 tile
        const float* src = Y + ((unsigned long long)b * NC) * HW + h0 + t;
        float sc = 0.f, sq = 0.f;
        #pragma unroll 8
        for (int c = 0; c < NC; ++c) {
            float v = src[(unsigned long long)c * HW];
            sc += v;
            sq = fmaf(v, v, sq);
            s_lds[c * HC + t] += v;
            ybf[c * LDSPITCH + t] = f2bf_rne(v);
        }
        float mean = sc * (1.f / 64.f);
        float var  = (sq - 64.f * mean * mean) * (1.f / 63.f);
        float r    = 1.f - sqrtf(var + 1e-4f);
        stdacc += fmaxf(r, 0.f);
        __syncthreads();

        // ---- phase 2: Gram accumulation; wave w does tiles (ci=w, di=0..3)
        #pragma unroll
        for (int ks = 0; ks < HC / 32; ++ks) {
            v16bf a = load_fragA(ybf, w, ks, lane);
            #pragma unroll
            for (int di = 0; di < 4; ++di) {
                v16bf bb = load_fragB(ybf, di, ks, lane);
                accY[di] = __builtin_amdgcn_wmma_f32_16x16x32_bf16(
                    false, a, false, bb, (short)0, accY[di], false, false);
            }
        }
        __syncthreads();
    }

    // ---- s-Gram: tile = s/8 in bf16, accumulate (s/8)(s/8)^T = (1/B) s s^T
    for (int f = t; f < NC * HC; f += 128) {
        int c = f >> 7, h = f & (HC - 1);
        ybf[c * LDSPITCH + h] = f2bf_rne(s_lds[c * HC + h] * 0.125f);
    }
    __syncthreads();
    #pragma unroll
    for (int ks = 0; ks < HC / 32; ++ks) {
        v16bf a = load_fragA(ybf, w, ks, lane);
        #pragma unroll
        for (int di = 0; di < 4; ++di) {
            v16bf bb = load_fragB(ybf, di, ks, lane);
            accS[di] = __builtin_amdgcn_wmma_f32_16x16x32_bf16(
                false, a, false, bb, (short)0, accS[di], false, false);
        }
    }

    // ---- writeout: G += accY - accS  (C/D layout: VGPR j, lane l -> M=j+8*(l>>4), N=l&15)
    #pragma unroll
    for (int di = 0; di < 4; ++di) {
        #pragma unroll
        for (int j = 0; j < 8; ++j) {
            int M = j + ((lane >> 4) << 3);
            int N = lane & 15;
            atomicAdd(&gG[(w * 16 + M) * 64 + di * 16 + N], accY[di][j] - accS[di][j]);
        }
    }

    // ---- std partial reduction
    red[t] = stdacc;
    __syncthreads();
    for (int sft = 64; sft; sft >>= 1) {
        if (t < sft) red[t] += red[t + sft];
        __syncthreads();
    }
    if (t == 0) atomicAdd(gStd, red[0]);
}

__global__ void vcreg_finalize(const float* __restrict__ gG, const float* __restrict__ gStd,
                               float* __restrict__ out, unsigned long long HW) {
    __shared__ float red[64];
    int t = threadIdx.x;
    float inv = 1.f / ((float)HW * 63.f);
    float off = 0.f;
    for (int i = t; i < 4096; i += 64) {
        int rr = i >> 6, cc = i & 63;
        float cv = gG[i] * inv;
        if (rr != cc) off = fmaf(cv, cv, off);
    }
    red[t] = off;
    __syncthreads();
    for (int sft = 32; sft; sft >>= 1) {
        if (t < sft) red[t] += red[t + sft];
        __syncthreads();
    }
    if (t == 0) {
        float cov_loss = red[0] * (1.f / 64.f);
        float std_loss = gStd[0] / (64.f * (float)HW);
        out[0] = 25.f * std_loss + cov_loss;
    }
}

__global__ void vcreg_zero(float* __restrict__ p, int n) {
    int i = blockIdx.x * blockDim.x + threadIdx.x;
    if (i < n) p[i] = 0.f;
}

extern "C" void kernel_launch(void* const* d_in, const int* in_sizes, int n_in,
                              void* d_out, int out_size, void* d_ws, size_t ws_size,
                              hipStream_t stream) {
    const float* Y   = (const float*)d_in[0];
    float*       out = (float*)d_out;
    float*       gG  = (float*)d_ws;        // [64*64] raw Gram (yy^T - ss^T/B)
    float*       gStd = gG + 4096;          // [1] sum of relu(1-std)

    unsigned long long HW = (unsigned long long)in_sizes[0] / (NB * NC);  // 50176
    int nchunks = (int)(HW / HC);                                          // 392

    vcreg_zero<<<(4097 + 255) / 256, 256, 0, stream>>>(gG, 4097);
    vcreg_main<<<nchunks, 128, 0, stream>>>(Y, HW, gG, gStd);
    vcreg_finalize<<<1, 64, 0, stream>>>(gG, gStd, out, HW);
}
